// CgpHmmCell_89610197664187
// MI455X (gfx1250) — matrix operand order repaired and checked by reading, not compiled, not run on hardware
//
#include <hip/hip_runtime.h>
#include <hip/hip_bf16.h>
#include <math.h>

// Problem dims (from reference)
#define BATCH 32
#define TLEN  4096
#define NS    512
#define NE    128
#define NWG   16          // persistent workgroups, each owns 32 state columns
#define WGCOLS 32
#define R_EPS 1e-12f
#define E_EPS 1e-10f
#define L_EPS 1e-10f

// LDS row strides (padded so row stride != 0 mod 64 banks)
#define AST 516           // alpha f32 row stride (dwords): 512 + 2 pad every 256 (TDM pad)
#define PST 520           // P f16 row stride (halves)

typedef _Float16 v16h __attribute__((ext_vector_type(16)));
typedef _Float16 v8h  __attribute__((ext_vector_type(8)));
typedef float    v8f  __attribute__((ext_vector_type(8)));
typedef unsigned int v4u __attribute__((ext_vector_type(4)));
typedef int      v8i  __attribute__((ext_vector_type(8)));
typedef int      v4i  __attribute__((ext_vector_type(4)));

__device__ __forceinline__ unsigned lds_off(const void* p) {
  // flat LDS aperture address: low 32 bits are the wave-relative LDS byte offset
  return (unsigned)(unsigned long long)p;
}

// Tensor Data Mover: 2D tile (global -> LDS), 6-arg builtin signature.
// flags_w0 packs data_size / pad controls into D# group1 word 0.
__device__ __forceinline__ void tdm_load_2d(unsigned ldsoff, const void* gsrc,
                                            unsigned flags_w0,
                                            unsigned dim0, unsigned dim1,
                                            unsigned tile0, unsigned tile1,
                                            unsigned stride0) {
  unsigned long long ga = (unsigned long long)gsrc;
  v4u g0;
  g0.x = 1u;                                              // count=1, user descriptor
  g0.y = ldsoff;                                          // lds_addr
  g0.z = (unsigned)ga;                                    // global_addr[31:0]
  g0.w = (unsigned)((ga >> 32) & 0x01ffffffu) | (2u << 30); // global_addr[56:32] | type=2
  v8i g1;
  g1[0] = (int)flags_w0;                                  // wg_mask=0 | data_size | pad ctl
  g1[1] = (int)((dim0 & 0xffffu) << 16);                  // tensor_dim0[15:0] @ [31:16]
  g1[2] = (int)((dim0 >> 16) | ((dim1 & 0xffffu) << 16)); // dim0 hi | dim1 lo
  g1[3] = (int)((dim1 >> 16) | (tile0 << 16));            // dim1 hi | tile_dim0
  g1[4] = (int)(tile1 & 0xffffu);                         // tile_dim1 (tile_dim2=0)
  g1[5] = (int)stride0;                                   // tensor_dim0_stride[31:0]
  g1[6] = 0;
  g1[7] = 0;
  v4i z4 = {0, 0, 0, 0};
  v8i z8 = {0, 0, 0, 0, 0, 0, 0, 0};
  __builtin_amdgcn_tensor_load_to_lds(g0, g1, z4, z4, z8, 0);
}

// Device-wide barrier: cluster barrier when dispatched as a cluster,
// sense-reversing L2 atomic barrier otherwise (16 co-resident WGs).
__device__ __forceinline__ void grid_sync(unsigned* bar) {
  __threadfence();
  __syncthreads();
  if (__builtin_amdgcn_cluster_id_x() != 0) {
    __builtin_amdgcn_s_cluster_barrier();
  } else if (threadIdx.x == 0) {
    unsigned g = __hip_atomic_load(&bar[1], __ATOMIC_RELAXED, __HIP_MEMORY_SCOPE_AGENT);
    unsigned a = __hip_atomic_fetch_add(&bar[0], 1u, __ATOMIC_ACQ_REL, __HIP_MEMORY_SCOPE_AGENT);
    if (a == NWG - 1) {
      __hip_atomic_store(&bar[0], 0u, __ATOMIC_RELAXED, __HIP_MEMORY_SCOPE_AGENT);
      __hip_atomic_fetch_add(&bar[1], 1u, __ATOMIC_RELEASE, __HIP_MEMORY_SCOPE_AGENT);
    } else {
      // Acquire-load poll: L2 round trip paces the spin.
      while (__hip_atomic_load(&bar[1], __ATOMIC_ACQUIRE, __HIP_MEMORY_SCOPE_AGENT) == g)
        ;
    }
  }
  __syncthreads();
}

// ---------------- pre-kernels ----------------

__global__ void init_barrier_k(unsigned* bar) { bar[0] = 0u; bar[1] = 0u; }

// A = softmax(A_kernel, axis=1) -> f16 [512][512]
__global__ void trans_softmax_f16(const float* __restrict__ Ak, _Float16* __restrict__ Af16) {
  __shared__ float red[256];
  const int row = blockIdx.x;
  const int t = threadIdx.x;
  const float* r = Ak + row * NS;
  float v0 = r[t], v1 = r[t + 256];
  red[t] = fmaxf(v0, v1);
  __syncthreads();
  for (int s = 128; s > 0; s >>= 1) { if (t < s) red[t] = fmaxf(red[t], red[t + s]); __syncthreads(); }
  float mx = red[0];
  __syncthreads();
  float e0 = __expf(v0 - mx), e1 = __expf(v1 - mx);
  red[t] = e0 + e1;
  __syncthreads();
  for (int s = 128; s > 0; s >>= 1) { if (t < s) red[t] += red[t + s]; __syncthreads(); }
  float inv = 1.0f / red[0];
  Af16[row * NS + t]       = (_Float16)(e0 * inv);
  Af16[row * NS + t + 256] = (_Float16)(e1 * inv);
}

// logBmE = log(softmax(B_kernel, axis=0) + E_EPS)  [128][512]
__global__ void emis_logsoftmax(const float* __restrict__ Bk, float* __restrict__ logBmE) {
  __shared__ float red[128];
  const int col = blockIdx.x;
  const int e = threadIdx.x;
  float v = Bk[e * NS + col];
  red[e] = v;
  __syncthreads();
  for (int s = 64; s > 0; s >>= 1) { if (e < s) red[e] = fmaxf(red[e], red[e + s]); __syncthreads(); }
  float mx = red[0];
  __syncthreads();
  float ex = __expf(v - mx);
  red[e] = ex;
  __syncthreads();
  for (int s = 64; s > 0; s >>= 1) { if (e < s) red[e] += red[e + s]; __syncthreads(); }
  float p = ex / red[0];
  logBmE[e * NS + col] = __logf(p + E_EPS);
}

// obs[b,t] = argmax_e onehot[b,t,e]  (one-hot -> exactly one element > 0.5)
__global__ void obs_argmax(const float* __restrict__ onehot, int* __restrict__ obs) {
  long long bt = blockIdx.x;
  float v = onehot[bt * NE + threadIdx.x];
  if (v > 0.5f) obs[bt] = threadIdx.x;
}

// ---------------- main persistent scan kernel ----------------

__global__ void __launch_bounds__(256, 1)
hmm_forward_main(const _Float16* __restrict__ Af16,   // [512][512] softmax(A) in f16
                 const float* __restrict__ logBmE,    // [128][512] log emissions
                 const int* __restrict__ obs,         // [32][4096]
                 float* __restrict__ G,               // double-buffered alpha [2][32][512]
                 float* __restrict__ out,             // [32] loglik
                 unsigned* __restrict__ bar) {
  extern __shared__ char smem[];
  float*    alphaS = (float*)smem;                               // 32 x AST f32 (TDM-padded)
  _Float16* Ps     = (_Float16*)(smem + 32 * AST * 4);           // 32 x PST f16
  float*    Bms    = (float*)(smem + 32 * AST * 4 + 32 * PST * 2); // 128 x 32 f32
  float*    mS     = Bms + NE * WGCOLS;                          // 32
  float*    pm     = mS + 32;                                    // 32 x 8 partials
  int*      obsS   = (int*)(pm + 32 * 8);                        // 32

  const int tid = threadIdx.x;
  const int wg  = blockIdx.x;
  const int wv  = tid >> 5;
  const int ln  = tid & 31;
  const int nbase = wg * WGCOLS;

  // ---- one-time: emission slice via TDM (128 x 32 tile of logBmE) ----
  if (wv == 0) {
    tdm_load_2d(lds_off(Bms), logBmE + nbase, /*data_size=4B*/ 2u << 16,
                /*dim0*/ NS, /*dim1*/ NE, /*tile0*/ WGCOLS, /*tile1*/ NE, /*stride*/ NS);
    __builtin_amdgcn_s_wait_tensorcnt(0);
  }
  __syncthreads();

  // ---- one-time: WMMA B-fragments. 8 waves: wave pair (w, w+4) shares one
  // 16x16 output tile, splitting K (w: K 0..255, w+4: K 256..511).
  const int mt = wv & 1;          // batch-row tile (0..1)
  const int nt = (wv >> 1) & 1;   // 16-col subtile (0..1)
  const int ks = wv >> 2;         // K half (0..1)
  const int n16 = ln & 15;
  const int kh  = ln >> 4;
  v16h bfrag[8];
#pragma unroll
  for (int j = 0; j < 8; ++j) {
    const int kk = ks * 8 + j;
#pragma unroll
    for (int i = 0; i < 16; ++i)
      bfrag[j][i] = Af16[(kk * 32 + kh * 16 + i) * NS + nbase + nt * 16 + n16];
  }

  // ---- init: alpha0 = logE0 + [0, -inf...] into G[0] ----
  {
    int b  = tid >> 3;
    int c0 = (tid & 7) * 4;
    int ob = obs[b * TLEN + 0];
    for (int i = 0; i < 4; ++i) {
      int cl = c0 + i;
      int gc = nbase + cl;
      G[b * NS + gc] = Bms[ob * WGCOLS + cl] + (gc == 0 ? 0.0f : -1e30f);
    }
  }
  grid_sync(bar);

  // alpha TDM flags: data_size=4B, pad_enable, pad every 256 dwords by 2 dwords
  const unsigned aw0 = (2u << 16) | (1u << 20) | (7u << 22) | (1u << 25);

  for (int t = 1; t < TLEN; ++t) {
    const float* src = G + ((t - 1) & 1) * (BATCH * NS);
    float*       dst = G + (t & 1) * (BATCH * NS);

    // Phase A: async-DMA full alpha [32][512] into padded LDS (one TDM instr)
    if (wv == 0) {
      tdm_load_2d(lds_off(alphaS), src, aw0, NS, BATCH, NS, BATCH, NS);
      __builtin_amdgcn_s_wait_tensorcnt(0);
    }
    if (tid < BATCH) obsS[tid] = obs[tid * TLEN + t];
    __syncthreads();

    // Phase B: row max m[b] (32 rows x 8 chunks of 64)
    {
      int b = tid & 31, ch = tid >> 5;
      int c = ch * 64;
      const float* p = alphaS + b * AST + c + ((c >> 8) << 1);  // skip mid-row TDM pad
      float mx = -3.0e38f;
      for (int i = 0; i < 64; ++i) mx = fmaxf(mx, p[i]);
      pm[b * 8 + ch] = mx;
    }
    __syncthreads();
    if (tid < 32) {
      float mx = pm[tid * 8];
      for (int i = 1; i < 8; ++i) mx = fmaxf(mx, pm[tid * 8 + i]);
      mS[tid] = mx;
    }
    __syncthreads();

    // Phase C: P = exp(alpha - m) + R_EPS  -> f16 (padded rows)
    {
      int b = tid & 31, ch = tid >> 5;
      float m = mS[b];
      int c = ch * 64;
      const float* p = alphaS + b * AST + c + ((c >> 8) << 1);
      _Float16* q = Ps + b * PST + c;
      for (int i = 0; i < 64; ++i)
        q[i] = (_Float16)(__expf(p[i] - m) + R_EPS);
    }
    __syncthreads();

    // Phase D: all 8 waves do WMMA (2 per SIMD interleave to hide ds latency).
    // Each wave: preload its 8 a-frags, then 8 chained WMMAs over its K half.
    {
      const _Float16* Prow = Ps + (mt * 16 + n16) * PST;
      v16h afr[8];
#pragma unroll
      for (int j = 0; j < 8; ++j) {
        const int kk = ks * 8 + j;
        v8h lo = *(const v8h*)(Prow + kk * 32 + kh * 16);
        v8h hi = *(const v8h*)(Prow + kk * 32 + kh * 16 + 8);
#pragma unroll
        for (int i = 0; i < 8; ++i) { afr[j][i] = lo[i]; afr[j][8 + i] = hi[i]; }
      }
      v8f acc = {0.f, 0.f, 0.f, 0.f, 0.f, 0.f, 0.f, 0.f};
#pragma unroll
      for (int j = 0; j < 8; ++j)
        acc = __builtin_amdgcn_wmma_f32_16x16x32_f16(false, afr[j], false, bfrag[j],
                                                     (short)0, acc, false, false);

      // Combine K halves: waves 4..7 dump partials into (now dead) alphaS scratch.
      float* part = alphaS;
      if (ks == 1) {
#pragma unroll
        for (int r = 0; r < 8; ++r) part[((wv - 4) * 32 + ln) * 8 + r] = acc[r];
      }
      __syncthreads();
      if (ks == 0) {
        const float* q = part + (wv * 32 + ln) * 8;
        const int gcol = nbase + nt * 16 + n16;
#pragma unroll
        for (int r = 0; r < 8; ++r) {
          float c = acc[r] + q[r];
          int brow = mt * 16 + kh * 8 + r;
          float R  = __logf(c) + mS[brow];
          dst[brow * NS + gcol] = Bms[obsS[brow] * WGCOLS + nt * 16 + n16] + R;
        }
      }
    }
    grid_sync(bar);
  }

  // ---- final loglik from G[(TLEN-1)&1] ----
  if (wg == 0) {
    const float* fin = G + ((TLEN - 1) & 1) * (BATCH * NS);
    int b = tid & 31, ch = tid >> 5;
    const float* row = fin + b * NS + ch * 64;
    float mx = -3.0e38f;
    for (int i = 0; i < 64; ++i) mx = fmaxf(mx, row[i]);
    pm[b * 8 + ch] = mx;
    __syncthreads();
    if (tid < 32) {
      float m2 = pm[tid * 8];
      for (int i = 1; i < 8; ++i) m2 = fmaxf(m2, pm[tid * 8 + i]);
      mS[tid] = m2;
    }
    __syncthreads();
    float m2 = mS[b];
    float s = 0.f;
    for (int i = 0; i < 64; ++i) s += __expf(row[i] - m2) + L_EPS;
    pm[b * 8 + ch] = s;
    __syncthreads();
    if (tid < 32) {
      float s2 = 0.f;
      for (int i = 0; i < 8; ++i) s2 += pm[tid * 8 + i];
      out[tid] = __logf(s2) + mS[tid];
    }
  }
}

// ---------------- host launcher ----------------

extern "C" void kernel_launch(void* const* d_in, const int* in_sizes, int n_in,
                              void* d_out, int out_size, void* d_ws, size_t ws_size,
                              hipStream_t stream) {
  const float* inputs = (const float*)d_in[0];   // [32,4096,128] one-hot f32
  const float* Ak     = (const float*)d_in[1];   // [512,512]
  const float* Bk     = (const float*)d_in[2];   // [128,512]

  char* ws = (char*)d_ws;
  _Float16* Af16  = (_Float16*)(ws);                 // 512 KB
  float*    logBm = (float*)(ws + (512 << 10));      // 256 KB
  int*      obs   = (int*)(ws + (768 << 10));        // 512 KB
  float*    G     = (float*)(ws + (1280 << 10));     // 128 KB (double-buffered alpha)
  unsigned* bar   = (unsigned*)(ws + (1408 << 10));  // 8 B

  hipLaunchKernelGGL(init_barrier_k, dim3(1), dim3(1), 0, stream, bar);
  hipLaunchKernelGGL(trans_softmax_f16, dim3(NS), dim3(256), 0, stream, Ak, Af16);
  hipLaunchKernelGGL(emis_logsoftmax, dim3(NS), dim3(NE), 0, stream, Bk, logBm);
  hipLaunchKernelGGL(obs_argmax, dim3(BATCH * TLEN), dim3(NE), 0, stream, inputs, obs);

  size_t smem = (size_t)32 * AST * 4   // alpha staging
              + (size_t)32 * PST * 2   // P f16
              + (size_t)NE * WGCOLS * 4// emission slice
              + 32 * 4                 // m
              + 32 * 8 * 4             // partials
              + 32 * 4;                // obs_t
  hipLaunchKernelGGL(hmm_forward_main, dim3(NWG), dim3(256), smem, stream,
                     Af16, logBm, obs, G, (float*)d_out, bar);
}
